// GCL_skip_global_28681791603391
// MI455X (gfx1250) — matrix operand order repaired and checked by reading, not compiled, not run on hardware
//
#include <hip/hip_runtime.h>
#include <hip/hip_bf16.h>

#define N_NODES 50000
#define DIM     512
#define NEDGE   500000

typedef __attribute__((ext_vector_type(16))) __bf16 v16bf;
typedef __attribute__((ext_vector_type(8)))  float  v8f;

static __device__ __forceinline__ unsigned short f32_bf16(float f) {
    unsigned int u = __float_as_uint(f);
    u += 0x7FFFu + ((u >> 16) & 1u);           // round-to-nearest-even
    return (unsigned short)(u >> 16);
}

// ---------------------------------------------------------------------------
// Kernel 1: convert the three 512x512 f32 weight matrices to bf16, laid out in
// this file's contiguous-k per-lane B-fragment order (same as the A reads):
//   B (32x16, 16-bit): lanes 0-15 hold col n, K=0..15 (VGPR v -> K=2v,2v+1),
//   lanes 16-31 hold K=16..31.  Stored as [g][k0][nt][lane][16].
// ---------------------------------------------------------------------------
__global__ void k_prep_weights(const float* __restrict__ wh,
                               const float* __restrict__ wsm,
                               const float* __restrict__ wmm,
                               unsigned short* __restrict__ wsw) {
    int idx = blockIdx.x * blockDim.x + threadIdx.x;   // 0 .. 3*512*512-1
    int g   = idx >> 18;
    int rem = idx & (DIM * DIM - 1);
    int k = rem >> 9;
    int n = rem & 511;
    const float* w = (g == 0) ? wh : (g == 1) ? wsm : wmm;
    float f = w[rem];
    int k0 = k >> 5, kk = k & 31;
    int nt = n >> 4, nn = n & 15;
    int lane = nn + ((kk & 16) ? 16 : 0);
    int pos  = kk & 15;
    size_t off = ((((size_t)g * 16 + k0) * 32 + nt) * 32 + lane) * 16 + pos;
    wsw[off] = f32_bf16(f);
}

__global__ void k_bias(const float* __restrict__ bh, const float* __restrict__ bs,
                       const float* __restrict__ bm, float* __restrict__ bsum) {
    int i = blockIdx.x * blockDim.x + threadIdx.x;
    if (i < DIM) bsum[i] = bh[i] + bs[i] + bm[i];
}

// ---------------------------------------------------------------------------
// Kernel 2: WMMA GEMM  out[N,512] = act[N,512] @ W[512,512]  (bf16 x bf16 -> f32)
// Block: 256 threads = 8 waves as 2(M) x 4(N); block tile 32 x 512.
//  - A: f32 -> bf16 through double-buffered LDS (one barrier per k-step),
//    next k-step's global float4 prefetched during compute.
//  - B: pre-swizzled bf16, all 8 fragments (16 x global_load_b128) issued
//    before the barrier into distinct registers so the 8 WMMAs don't stall
//    on a full loadcnt wait each.
// ---------------------------------------------------------------------------
__global__ __launch_bounds__(256)
void k_gemm(const float* __restrict__ h, const float* __restrict__ s,
            const float* __restrict__ m,
            const float* __restrict__ norm_g, const float* __restrict__ norm_f,
            const unsigned short* __restrict__ wsw,
            const float* __restrict__ bsum,
            float* __restrict__ ph, float* __restrict__ ps,
            float* __restrict__ base)
{
    const int g = blockIdx.y;
    const float* act = (g == 0) ? h : (g == 1) ? s : m;
    float* outp      = (g == 0) ? ph : (g == 1) ? ps : base;
    const unsigned short* w = wsw + (size_t)g * (16 * 32 * 32 * 16);

    __shared__ unsigned short alds[2][32 * 40];  // double-buffered A tile

    const int tid  = threadIdx.x;
    const int lane = tid & 31;
    const int wave = tid >> 5;
    const int wm_  = wave >> 2;     // 0..1 : which 16-row half
    const int wn   = wave & 3;      // 0..3 : which 128-col quarter
    const int half = lane >> 4;
    const int l15  = lane & 15;

    const int row0 = blockIdx.x * 32;

    v8f acc[8];
    v8f vzero = {0.f, 0.f, 0.f, 0.f, 0.f, 0.f, 0.f, 0.f};
#pragma unroll
    for (int t = 0; t < 8; ++t) acc[t] = vzero;

    const int a_r = tid >> 3;          // 0..31
    const int a_c = (tid & 7) * 4;     // 0,4,..,28
    const int agrow = row0 + a_r;
    const float* arow = act + (size_t)agrow * DIM + a_c;

    // prime the A pipeline
    float4 av = make_float4(0.f, 0.f, 0.f, 0.f);
    if (agrow < N_NODES) av = *(const float4*)(arow);

    int buf = 0;
    for (int k0 = 0; k0 < 16; ++k0) {
        // --- issue all 8 B fragments for this k-step (independent of LDS)
        union { uint4 q[2]; v16bf v; } bfr[8];
        const unsigned short* wk = w + (size_t)k0 * (32 * 32 * 16);
#pragma unroll
        for (int t = 0; t < 8; ++t) {
            const unsigned short* bp = wk + ((size_t)(wn * 8 + t) * 32 + lane) * 16;
            bfr[t].q[0] = *(const uint4*)(bp);
            bfr[t].q[1] = *(const uint4*)(bp + 8);
        }

        // --- store current A tile (bf16) into LDS buffer `buf`
        unsigned int p0 = (unsigned)f32_bf16(av.x) | ((unsigned)f32_bf16(av.y) << 16);
        unsigned int p1 = (unsigned)f32_bf16(av.z) | ((unsigned)f32_bf16(av.w) << 16);
        *(uint2*)&alds[buf][a_r * 40 + a_c] = make_uint2(p0, p1);

        // --- prefetch next k-step's A while this one is consumed
        if (k0 < 15) {
            av = make_float4(0.f, 0.f, 0.f, 0.f);
            if (agrow < N_NODES) av = *(const float4*)(arow + (k0 + 1) * 32);
        }

        __syncthreads();   // single barrier per k-step (double buffer)

        // --- A fragment (contiguous k order chosen here, not AMD's table order):
        //     lane half 0: K {0..15}; half 1: K {16..31} (matches B packing)
        union { uint2 d[4]; v16bf v; } af;
        const unsigned short* ap = &alds[buf][(wm_ * 16 + l15) * 40];
        af.d[0] = *(const uint2*)(ap + half * 16);
        af.d[1] = *(const uint2*)(ap + half * 16 + 4);
        af.d[2] = *(const uint2*)(ap + half * 16 + 8);
        af.d[3] = *(const uint2*)(ap + half * 16 + 12);

        // --- 8 WMMAs; B data already in (distinct) registers
#pragma unroll
        for (int t = 0; t < 8; ++t) {
            acc[t] = __builtin_amdgcn_wmma_f32_16x16x32_bf16(
                false, af.v, false, bfr[t].v, (short)0, acc[t], false, false);
        }
        buf ^= 1;
    }

    // --- epilogue: C/D layout: VGPR r, lanes 0-15 -> M=r, lanes 16-31 -> M=r+8
#pragma unroll
    for (int r = 0; r < 8; ++r) {
        int grow = row0 + wm_ * 16 + half * 8 + r;
        if (grow >= N_NODES) continue;
        float rscale = (g == 0) ? norm_g[grow] : (g == 1) ? norm_f[grow] : 1.0f;
#pragma unroll
        for (int t = 0; t < 8; ++t) {
            int col = wn * 128 + t * 16 + l15;
            float v = acc[t][r] * rscale;
            if (g == 2) v += bsum[col];
            outp[(size_t)grow * DIM + col] = v;
        }
    }
}

// ---------------------------------------------------------------------------
// Kernel 3: edge scatter-add.  One wave per edge; 512 floats per row.
// Fold norm[dst] into the contribution so out accumulates the final term.
// ---------------------------------------------------------------------------
__global__ __launch_bounds__(256)
void k_edges(const float* __restrict__ P,
             const int* __restrict__ srcs, const int* __restrict__ dsts,
             const float* __restrict__ nrm,
             float* __restrict__ out)
{
    int e = blockIdx.x * 8 + (threadIdx.x >> 5);
    if (e >= NEDGE) return;
    int lane = threadIdx.x & 31;
    int src = srcs[e];
    int dst = dsts[e];
    float sc = nrm[dst];
    const float* pr = P + (size_t)src * DIM;
    float* orow = out + (size_t)dst * DIM;
#pragma unroll
    for (int j = 0; j < 4; ++j) {
        int c = j * 128 + lane * 4;
        float4 v = *(const float4*)(pr + c);
        unsafeAtomicAdd(orow + c + 0, v.x * sc);
        unsafeAtomicAdd(orow + c + 1, v.y * sc);
        unsafeAtomicAdd(orow + c + 2, v.z * sc);
        unsafeAtomicAdd(orow + c + 3, v.w * sc);
    }
}

__global__ void k_relu(float* __restrict__ out, int n4) {
    int i = blockIdx.x * blockDim.x + threadIdx.x;
    if (i < n4) {
        float4* p = (float4*)out + i;
        float4 v = *p;
        v.x = fmaxf(v.x, 0.f);
        v.y = fmaxf(v.y, 0.f);
        v.z = fmaxf(v.z, 0.f);
        v.w = fmaxf(v.w, 0.f);
        *p = v;
    }
}

// ---------------------------------------------------------------------------
extern "C" void kernel_launch(void* const* d_in, const int* in_sizes, int n_in,
                              void* d_out, int out_size, void* d_ws, size_t ws_size,
                              hipStream_t stream)
{
    (void)in_sizes; (void)n_in; (void)out_size; (void)ws_size;

    const float* h      = (const float*)d_in[0];
    const float* s      = (const float*)d_in[1];
    const float* m      = (const float*)d_in[2];
    const float* norm_g = (const float*)d_in[3];
    const float* norm_f = (const float*)d_in[4];
    const int*   src_g  = (const int*)d_in[5];
    const int*   dst_g  = (const int*)d_in[6];
    const int*   src_f  = (const int*)d_in[7];
    const int*   dst_f  = (const int*)d_in[8];
    const float* wh     = (const float*)d_in[9];
    const float* wsm    = (const float*)d_in[10];
    const float* wmm    = (const float*)d_in[11];
    const float* bh     = (const float*)d_in[12];
    const float* bs     = (const float*)d_in[13];
    const float* bm     = (const float*)d_in[14];

    char* wp = (char*)d_ws;
    unsigned short* wsw = (unsigned short*)wp;                 // 1.5 MB
    float* bsum = (float*)(wp + (size_t)3 * DIM * DIM * 2);    // 2 KB (pad to 4K)
    float* ph   = (float*)(wp + (size_t)3 * DIM * DIM * 2 + 4096);
    float* ps   = ph + (size_t)N_NODES * DIM;
    float* base = (float*)d_out;

    k_prep_weights<<<(3 * DIM * DIM) / 256, 256, 0, stream>>>(wh, wsm, wmm, wsw);
    k_bias<<<2, 256, 0, stream>>>(bh, bs, bm, bsum);

    dim3 gg((N_NODES + 31) / 32, 3);
    k_gemm<<<gg, 256, 0, stream>>>(h, s, m, norm_g, norm_f, wsw, bsum, ph, ps, base);

    // Two sequential edge phases so each phase's (P + out) working set
    // (~204 MB) approximately fits the 192 MB L2.
    k_edges<<<(NEDGE + 7) / 8, 256, 0, stream>>>(ph, src_g, dst_g, norm_g, base);
    k_edges<<<(NEDGE + 7) / 8, 256, 0, stream>>>(ps, src_f, dst_f, norm_f, base);

    int n4 = N_NODES * DIM / 4;
    k_relu<<<(n4 + 255) / 256, 256, 0, stream>>>(base, n4);
}
